// TestILE_91190745629191
// MI455X (gfx1250) — matrix operand order repaired and checked by reading, not compiled, not run on hardware
//
#include <hip/hip_runtime.h>
#include <hip/hip_bf16.h>
#include <cstddef>

// ---------------------------------------------------------------------------
// Problem constants (from reference)
// ---------------------------------------------------------------------------
#define Bd   128
#define Td   512
#define Hd   128
#define Kd_  16
#define Ad   12
#define Md   4
#define Fd   576
#define H2d  256
#define H4d  512
#define BTd  65536            // B*T
#define NEG_INF_F (-1e30f)

typedef __attribute__((ext_vector_type(16))) _Float16 v16h;
typedef __attribute__((ext_vector_type(8)))  float    v8f;

// ---------------------------------------------------------------------------
// WMMA fragment loaders (contiguous k order k = 16*(L>>4) + s, same in A and B, wave32)
// ---------------------------------------------------------------------------
// A-matrix 16x32 f16, row-major source (lda = row stride in elements).
// lane L: M = L&15 ; VGPR v holds K pair: k = ((L>>4)?16:0) + v*2
__device__ static inline v16h load_a_frag(const _Float16* __restrict__ a, int lda) {
  const int lane = threadIdx.x & 31;
  const int m = lane & 15;
  const int kh = (lane >> 4) << 4;          // 0 or 16
  v16h f;
#pragma unroll
  for (int v = 0; v < 8; ++v) {
    const int k = kh + (v << 1);
    f[2 * v]     = a[(size_t)m * lda + k];
    f[2 * v + 1] = a[(size_t)m * lda + k + 1];
  }
  return f;
}

// B-matrix 32x16, source is W16 (N x K, f16, row-major) used transposed: B[k][n]=W[n][k]
// lane L: N = L&15 ; kbase = (L>>4)*16 -> 16 contiguous halves per lane (2x b128)
__device__ static inline v16h load_b_frag_w16(const _Float16* __restrict__ w, int ldk) {
  const int lane = threadIdx.x & 31;
  const int n = lane & 15;
  const int kb = (lane >> 4) << 4;
  v16h f;
#pragma unroll
  for (int v = 0; v < 8; ++v) {
    const int k = kb + 2 * v;
    f[2 * v]     = w[(size_t)n * ldk + k];
    f[2 * v + 1] = w[(size_t)n * ldk + k + 1];
  }
  return f;
}

// B-matrix 32x16 f16, row-major K x N source (ldn = row stride)
__device__ static inline v16h load_b_frag_h(const _Float16* __restrict__ b, int ldn) {
  const int lane = threadIdx.x & 31;
  const int n = lane & 15;
  const int kb = (lane >> 4) << 4;
  v16h f;
#pragma unroll
  for (int v = 0; v < 8; ++v) {
    const int k = kb + 2 * v;
    f[2 * v]     = b[(size_t)k * ldn + n];
    f[2 * v + 1] = b[(size_t)(k + 1) * ldn + n];
  }
  return f;
}

__device__ static inline float sigmf(float x) { return 1.0f / (1.0f + expf(-x)); }

// Generic f32 -> f16 conversion (for weight matrices; one-time)
__global__ void cvt16_kernel(const float* __restrict__ src, _Float16* __restrict__ dst,
                             int n) {
  const int i = blockIdx.x * 256 + threadIdx.x;
  if (i < n) dst[i] = (_Float16)src[i];
}

// ---------------------------------------------------------------------------
// Conv 2x2 pad(1,1) over 4x5x5 -> 16x6x6, ReLU, output f16 (BT x 576)
// ---------------------------------------------------------------------------
__global__ void conv_kernel(const float* __restrict__ st, const float* __restrict__ cw,
                            const float* __restrict__ cb, _Float16* __restrict__ out) {
  const int i = blockIdx.x * 256 + threadIdx.x;
  if (i >= BTd * Fd) return;
  const int bt = i / Fd, f = i % Fd;
  const int oc = f / 36, rem = f % 36;
  const int oy = rem / 6, ox = rem % 6;
  const float* sp = st + (size_t)bt * 100;   // C*5*5
  float acc = cb[oc];
#pragma unroll
  for (int ic = 0; ic < 4; ++ic) {
#pragma unroll
    for (int ky = 0; ky < 2; ++ky) {
      const int iy = oy - 1 + ky;
      if (iy < 0 || iy >= 5) continue;
#pragma unroll
      for (int kx = 0; kx < 2; ++kx) {
        const int ix = ox - 1 + kx;
        if (ix < 0 || ix >= 5) continue;
        acc += sp[ic * 25 + iy * 5 + ix] * cw[((oc * 4 + ic) * 2 + ky) * 2 + kx];
      }
    }
  }
  out[i] = (_Float16)fmaxf(acc, 0.0f);
}

// ---------------------------------------------------------------------------
// Generic WMMA GEMM: C[M,N] = A[M,K](f16) @ W16[N,K]^T(f16) + bias (+bias2), opt ReLU
// Each wave computes a 16x64 strip (4 N-tiles share one A fragment).
// block = 128 threads (4 waves), grid = (N/64, M/64)
// ---------------------------------------------------------------------------
template <bool RELU, bool OUT16>
__global__ void gemm_kernel(const _Float16* __restrict__ A, const _Float16* __restrict__ W,
                            const float* __restrict__ bias, const float* __restrict__ bias2,
                            void* __restrict__ Cout, int Mrows, int N, int Kdim, int ldc) {
  const int lane = threadIdx.x & 31;
  const int wave = threadIdx.x >> 5;
  const int n0 = blockIdx.x * 64;
  const int m0 = (blockIdx.y * 4 + wave) * 16;
  if (m0 >= Mrows) return;                        // uniform per wave
  const _Float16* Arow = A + (size_t)m0 * Kdim;
  v8f acc[4] = {{}, {}, {}, {}};
  for (int k = 0; k < Kdim; k += 32) {
    const v16h af = load_a_frag(Arow + k, Kdim);
#pragma unroll
    for (int nt = 0; nt < 4; ++nt) {
      const v16h bf = load_b_frag_w16(W + (size_t)(n0 + nt * 16) * Kdim + k, Kdim);
      acc[nt] = __builtin_amdgcn_wmma_f32_16x16x32_f16(false, af, false, bf, (short)0,
                                                       acc[nt], false, false);
    }
  }
  const int rbase = m0 + ((lane >> 4) << 3);
#pragma unroll
  for (int nt = 0; nt < 4; ++nt) {
    const int col = n0 + nt * 16 + (lane & 15);
    float bv = bias ? bias[col] : 0.0f;
    if (bias2) bv += bias2[col];
#pragma unroll
    for (int r = 0; r < 8; ++r) {
      float v = acc[nt][r] + bv;
      if (RELU) v = fmaxf(v, 0.0f);
      const size_t o = (size_t)(rbase + r) * ldc + col;
      if (OUT16) ((_Float16*)Cout)[o] = (_Float16)v;
      else       ((float*)Cout)[o] = v;
    }
  }
}

// ---------------------------------------------------------------------------
// xproj GEMM with fragment-major output for the LSTM:
//   rows retiled b-major per time step: tile tm = t*8 + bb covers b = bb*16..+15
//   output Pout[tm][ntile(32)][lane(32)][r(8)]  (WMMA C-fragment layout, f32)
// grid = (512/64 = 8, (T*8)/4 = 1024), block = 128
// ---------------------------------------------------------------------------
__global__ void gemm_xproj_kernel(const _Float16* __restrict__ A,   // emb (BT x 256)
                                  const _Float16* __restrict__ W,   // w_ih f16 (512 x 256)
                                  const float* __restrict__ bias,
                                  const float* __restrict__ bias2,
                                  float* __restrict__ Pout) {
  const int lane = threadIdx.x & 31;
  const int wave = threadIdx.x >> 5;
  const int nt0 = blockIdx.x * 4;                  // base 16-wide N tile
  const int tm  = blockIdx.y * 4 + wave;           // t*8 + bb
  const int t = tm >> 3, bb = tm & 7;
  const int ldA = Td * H2d;                        // row stride between consecutive b
  const _Float16* Arow = A + ((size_t)(bb * 16) * Td + t) * H2d;
  v8f acc[4] = {{}, {}, {}, {}};
  for (int k = 0; k < H2d; k += 32) {
    const v16h af = load_a_frag(Arow + k, ldA);
#pragma unroll
    for (int nt = 0; nt < 4; ++nt) {
      const v16h bf = load_b_frag_w16(W + (size_t)(nt0 + nt) * 16 * H2d + k, H2d);
      acc[nt] = __builtin_amdgcn_wmma_f32_16x16x32_f16(false, af, false, bf, (short)0,
                                                       acc[nt], false, false);
    }
  }
#pragma unroll
  for (int nt = 0; nt < 4; ++nt) {
    const int ntile = nt0 + nt;
    const int col = ntile * 16 + (lane & 15);
    const float bv = bias[col] + bias2[col];
    float* op = Pout + (((size_t)tm * 32 + ntile) * 32 + lane) * 8;
#pragma unroll
    for (int r = 0; r < 8; ++r) op[r] = acc[nt][r] + bv;
  }
}

// ---------------------------------------------------------------------------
// Gather action embeddings into emb[:,128:256] (f16)
// ---------------------------------------------------------------------------
__global__ void act_gather_kernel(const int* __restrict__ actions,
                                  const float* __restrict__ act_emb,
                                  _Float16* __restrict__ emb) {
  const int i = blockIdx.x * 256 + threadIdx.x;
  if (i >= BTd * Hd) return;
  const int bt = i >> 7, j = i & 127;
  emb[(size_t)bt * H2d + Hd + j] = (_Float16)act_emb[actions[bt] * Hd + j];
}

// Softmax over A=12 within each (bt,k) group; in-place on (BT x 192) f32
__global__ void softmax12_kernel(float* __restrict__ p) {
  const int i = blockIdx.x * 256 + threadIdx.x;
  if (i >= BTd * Kd_) return;
  float* row = p + (size_t)(i >> 4) * (Kd_ * Ad) + (i & 15) * Ad;
  float mx = row[0];
#pragma unroll
  for (int a = 1; a < Ad; ++a) mx = fmaxf(mx, row[a]);
  float e[Ad], s = 0.0f;
#pragma unroll
  for (int a = 0; a < Ad; ++a) { e[a] = expf(row[a] - mx); s += e[a]; }
  const float inv = 1.0f / s;
#pragma unroll
  for (int a = 0; a < Ad; ++a) row[a] = e[a] * inv;
}

// Transpose w_hh (512x128 f32) -> whhT (128x512 f16): whhT[k*512+col] = w_hh[col*128+k]
__global__ void whh_tr_kernel(const float* __restrict__ whh, _Float16* __restrict__ whhT) {
  const int i = blockIdx.x * 256 + threadIdx.x;
  if (i >= Hd * H4d) return;
  const int k = i >> 9, col = i & 511;
  whhT[i] = (_Float16)whh[col * Hd + k];
}

__global__ void init_mask_kernel(float* __restrict__ logcum, float* __restrict__ mask) {
  const int i = blockIdx.x * 256 + threadIdx.x;
  if (i >= BTd) return;
  logcum[i] = 0.0f;
  mask[i] = 1.0f;
}

// ---------------------------------------------------------------------------
// LSTM recurrence for one segment. grid = 8 (batch tiles of 16), block = 256 (8 waves).
// wave w owns rows 16*bb..+15, hcols 16*w..+15, all 4 gates -> wave-local gate update.
// h double-buffered in LDS (f16); c state lives in VGPRs; whhT fragments are
// loop-invariant (compiler hoists them to VGPRs); xproj arrives in fragment-major
// layout so the gate init is 2x global_load_b128 per gate.
// ---------------------------------------------------------------------------
__global__ void __launch_bounds__(256) lstm_kernel(const float* __restrict__ xprojP,
                                                   const _Float16* __restrict__ whhT,
                                                   const float* __restrict__ maskp,
                                                   float* __restrict__ enc_out,
                                                   _Float16* __restrict__ enc16) {
  __shared__ _Float16 hbuf[2][16 * Hd];
  const int tid = threadIdx.x;
  const int bb = blockIdx.x;
  for (int i = tid; i < 2 * 16 * Hd; i += 256) ((_Float16*)hbuf)[i] = (_Float16)0.0f;
  __syncthreads();

  const int lane = tid & 31, wave = tid >> 5;
  const int hc0  = wave * 16;
  const int rlo  = (lane >> 4) << 3;     // 0 or 8 (C/D row offset)
  const int colL = lane & 15;
  float cst[8];
#pragma unroll
  for (int r = 0; r < 8; ++r) cst[r] = 0.0f;

  for (int t = 0; t < Td; ++t) {
    const _Float16* hcur = hbuf[t & 1];
    _Float16*       hnxt = hbuf[(t + 1) & 1];
    const v16h af0 = load_a_frag(hcur +  0, Hd);
    const v16h af1 = load_a_frag(hcur + 32, Hd);
    const v16h af2 = load_a_frag(hcur + 64, Hd);
    const v16h af3 = load_a_frag(hcur + 96, Hd);
    // fragment-major xproj base for (t, bb): [tm][ntile][lane][r]
    const float* xpt = xprojP + (((size_t)(t * 8 + bb) * 32) * 32) * 8;
    v8f g[4];
#pragma unroll
    for (int j = 0; j < 4; ++j) {
      const float* xp = xpt + (((size_t)(j * 8 + wave)) * 32 + lane) * 8;
      __builtin_prefetch(xp + 8 * 32 * 32 * 8, 0, 1);   // next time step (speculative)
      v8f acc;
#pragma unroll
      for (int r = 0; r < 8; ++r) acc[r] = xp[r];
      const _Float16* bp = whhT + j * Hd + hc0;
      acc = __builtin_amdgcn_wmma_f32_16x16x32_f16(false, af0, false,
              load_b_frag_h(bp,             H4d), (short)0, acc, false, false);
      acc = __builtin_amdgcn_wmma_f32_16x16x32_f16(false, af1, false,
              load_b_frag_h(bp + 32 * H4d,  H4d), (short)0, acc, false, false);
      acc = __builtin_amdgcn_wmma_f32_16x16x32_f16(false, af2, false,
              load_b_frag_h(bp + 64 * H4d,  H4d), (short)0, acc, false, false);
      acc = __builtin_amdgcn_wmma_f32_16x16x32_f16(false, af3, false,
              load_b_frag_h(bp + 96 * H4d,  H4d), (short)0, acc, false, false);
      g[j] = acc;
    }
#pragma unroll
    for (int r = 0; r < 8; ++r) {
      const int brow = rlo + r;
      const int b = bb * 16 + brow;
      const float m = maskp[(size_t)b * Td + t];
      const float iv = sigmf(g[0][r]);
      const float fv = sigmf(g[1][r]);
      const float gv = tanhf(g[2][r]);
      const float ov = sigmf(g[3][r]);
      float c = fv * cst[r] + iv * gv;
      float h = ov * tanhf(c);
      h *= m; c *= m;
      cst[r] = c;
      const int hcol = hc0 + colL;
      hnxt[brow * Hd + hcol] = (_Float16)h;
      const size_t eo = ((size_t)b * Td + t) * Hd + hcol;
      enc_out[eo] = h;
      enc16[eo]   = (_Float16)h;
    }
    __syncthreads();
  }
}

// logits_b = hid @ hb2_w.T + hb2_b, with t==0 forced to NEG_INF; written to lb output
__global__ void logitsb_kernel(const _Float16* __restrict__ hid,
                               const float* __restrict__ hb2_w,
                               const float* __restrict__ hb2_b,
                               float* __restrict__ lb_out) {
  const int i = blockIdx.x * 256 + threadIdx.x;
  if (i >= BTd) return;
  const int t = i & (Td - 1);
  float s = hb2_b[0];
  const _Float16* hp = hid + (size_t)i * Hd;
  for (int k = 0; k < Hd; ++k) s += (float)hp[k] * hb2_w[k];
  lb_out[i] = (t == 0) ? NEG_INF_F : s;
}

// First-occurrence argmax over T per batch row
__global__ void argmax_kernel(const float* __restrict__ logits, int* __restrict__ idx_out) {
  __shared__ float sv[256];
  __shared__ int   si[256];
  const int b = blockIdx.x, tid = threadIdx.x;
  float bv = -3.4e38f; int bi = 0x7fffffff;
  for (int t = tid; t < Td; t += 256) {
    const float v = logits[(size_t)b * Td + t];
    if (v > bv || (v == bv && t < bi)) { bv = v; bi = t; }
  }
  sv[tid] = bv; si[tid] = bi;
  __syncthreads();
  for (int s = 128; s > 0; s >>= 1) {
    if (tid < s) {
      const float ov = sv[tid + s]; const int oi = si[tid + s];
      if (ov > sv[tid] || (ov == sv[tid] && oi < si[tid])) { sv[tid] = ov; si[tid] = oi; }
    }
    __syncthreads();
  }
  if (tid == 0) idx_out[b] = si[0];
}

__global__ void lastidx_kernel(const int* __restrict__ lengths, int* __restrict__ idx) {
  const int b = threadIdx.x;
  if (b < Bd) idx[b] = lengths[b] - 1;
}

__global__ void sb_kernel(const int* __restrict__ idx, float* __restrict__ out) {
  const int i = blockIdx.x * 256 + threadIdx.x;
  if (i >= BTd) return;
  const int b = i >> 9, t = i & (Td - 1);
  out[i] = (t == idx[b]) ? 1.0f : 0.0f;
}

// readout[b,:] = (j>=1) ? enc[b, j-1, :] : 0   (sample_b is one-hot at j)
__global__ void readout_kernel(const int* __restrict__ idx, const float* __restrict__ enc,
                               float* __restrict__ ro) {
  const int b = blockIdx.x, h = threadIdx.x;
  const int j = idx[b];
  ro[b * Hd + h] = (j >= 1) ? enc[((size_t)b * Td + (j - 1)) * Hd + h] : 0.0f;
}

// logits_z = relu(ro @ hz1_w.T + hz1_b) @ hz2_w.T + hz2_b ; write lz (2K) and sz (K)
__global__ void logitsz_kernel(const float* __restrict__ ro, const float* __restrict__ hz1_w,
                               const float* __restrict__ hz1_b, const float* __restrict__ hz2_w,
                               const float* __restrict__ hz2_b, float* __restrict__ lz_out,
                               float* __restrict__ sz_out) {
  __shared__ float rsh[Hd];
  __shared__ float hsh[Hd];
  const int b = blockIdx.x, j = threadIdx.x;
  rsh[j] = ro[b * Hd + j];
  __syncthreads();
  float s = hz1_b[j];
  for (int k = 0; k < Hd; ++k) s += rsh[k] * hz1_w[j * Hd + k];
  hsh[j] = fmaxf(s, 0.0f);
  __syncthreads();
  if (j < 2 * Kd_) {
    float s2 = hz2_b[j];
    for (int k = 0; k < Hd; ++k) s2 += hsh[k] * hz2_w[j * Hd + k];
    lz_out[b * (2 * Kd_) + j] = s2;
    if (j < Kd_) sz_out[b * Kd_ + j] = s2;
  }
}

// mask update: cumsum(one_hot at j) = step(t>=j); log_cum += log(step+EPS); mask=exp
__global__ void maskupd_kernel(const int* __restrict__ idx, float* __restrict__ logcum,
                               float* __restrict__ mask, float* __restrict__ mask_out) {
  const int i = blockIdx.x * 256 + threadIdx.x;
  if (i >= BTd) return;
  const int b = i >> 9, t = i & (Td - 1);
  const float cum = (t >= idx[b]) ? 1.0f : 0.0f;
  const float lc = logcum[i] + logf(cum + 1e-17f);
  logcum[i] = lc;
  const float m = expf(lc);
  mask[i] = m;
  mask_out[i] = m;
}

// recs[bt,a] = sum_k sub_probs[bt,k,a] * sz[b,k]
__global__ void recs_kernel(const float* __restrict__ subp, const float* __restrict__ sz,
                            float* __restrict__ out) {
  const int i = blockIdx.x * 256 + threadIdx.x;
  if (i >= BTd * Ad) return;
  const int bt = i / Ad, a = i % Ad;
  const int b = bt >> 9;
  const float* p = subp + (size_t)bt * (Kd_ * Ad) + a;
  const float* z = sz + b * Kd_;
  float s = 0.0f;
#pragma unroll
  for (int k = 0; k < Kd_; ++k) s += p[k * Ad] * z[k];
  out[i] = s;
}

// ---------------------------------------------------------------------------
// Host launcher
// ---------------------------------------------------------------------------
extern "C" void kernel_launch(void* const* d_in, const int* in_sizes, int n_in,
                              void* d_out, int out_size, void* d_ws, size_t ws_size,
                              hipStream_t stream) {
  (void)in_sizes; (void)n_in; (void)out_size; (void)ws_size;
  const float* states  = (const float*)d_in[0];
  const int*   actions = (const int*)d_in[1];
  const int*   lengths = (const int*)d_in[2];
  const float* conv_w  = (const float*)d_in[3];
  const float* conv_b  = (const float*)d_in[4];
  const float* se_w1   = (const float*)d_in[5];
  const float* se_b1   = (const float*)d_in[6];
  const float* se_w2   = (const float*)d_in[7];
  const float* se_b2   = (const float*)d_in[8];
  const float* act_emb = (const float*)d_in[9];
  const float* w_ih    = (const float*)d_in[10];
  const float* w_hh    = (const float*)d_in[11];
  const float* b_ih    = (const float*)d_in[12];
  const float* b_hh    = (const float*)d_in[13];
  const float* hb1_w   = (const float*)d_in[14];
  const float* hb1_b   = (const float*)d_in[15];
  const float* hb2_w   = (const float*)d_in[16];
  const float* hb2_b   = (const float*)d_in[17];
  const float* hz1_w   = (const float*)d_in[18];
  const float* hz1_b   = (const float*)d_in[19];
  const float* hz2_w   = (const float*)d_in[20];
  const float* hz2_b   = (const float*)d_in[21];
  const float* dconv_w = (const float*)d_in[22];
  const float* dconv_b = (const float*)d_in[23];
  const float* sub_w   = (const float*)d_in[24];
  const float* sub_b   = (const float*)d_in[25];

  float* out = (float*)d_out;
  // output offsets (elements)
  const size_t OUT_ENCS  = 0;
  const size_t OUT_RECS  = OUT_ENCS  + (size_t)Md * BTd * Hd;
  const size_t OUT_MASKS = OUT_RECS  + (size_t)Md * BTd * Ad;
  const size_t OUT_LB    = OUT_MASKS + (size_t)(Md - 1) * BTd;
  const size_t OUT_SB    = OUT_LB    + (size_t)(Md - 1) * BTd;
  const size_t OUT_LZ    = OUT_SB    + (size_t)Md * BTd;
  const size_t OUT_SZ    = OUT_LZ    + (size_t)Md * Bd * 2 * Kd_;

  // workspace carve-up
  char* ws = (char*)d_ws;
  size_t off = 0;
  auto carve = [&](size_t bytes) { char* p = ws + off; off += (bytes + 255) & ~(size_t)255; return p; };
  _Float16* ws_conv   = (_Float16*)carve((size_t)BTd * Fd * 2);
  _Float16* ws_dec    = (_Float16*)carve((size_t)BTd * Fd * 2);
  _Float16* ws_hidden = (_Float16*)carve((size_t)BTd * Hd * 2);
  _Float16* ws_emb    = (_Float16*)carve((size_t)BTd * H2d * 2);
  float*    ws_xproj  = (float*)carve((size_t)BTd * H4d * 4);   // fragment-major
  float*    ws_sublog = (float*)carve((size_t)BTd * Kd_ * Ad * 4);
  _Float16* ws_enc16  = (_Float16*)carve((size_t)BTd * Hd * 2);
  _Float16* ws_hid    = (_Float16*)carve((size_t)BTd * Hd * 2);
  _Float16* ws_whhT   = (_Float16*)carve((size_t)Hd * H4d * 2);
  float*    ws_logcum = (float*)carve((size_t)BTd * 4);
  float*    ws_mask   = (float*)carve((size_t)BTd * 4);
  float*    ws_ro     = (float*)carve((size_t)Bd * Hd * 4);
  int*      ws_argmax = (int*)carve((size_t)Bd * 4);
  // f16 weight copies (one-time converts; keep B fragments contiguous & cvt-free)
  _Float16* w16_sew1  = (_Float16*)carve((size_t)Hd * Fd * 2);
  _Float16* w16_sew2  = (_Float16*)carve((size_t)Hd * Hd * 2);
  _Float16* w16_wih   = (_Float16*)carve((size_t)H4d * H2d * 2);
  _Float16* w16_subw  = (_Float16*)carve((size_t)Kd_ * Ad * Fd * 2);
  _Float16* w16_hb1   = (_Float16*)carve((size_t)Hd * Hd * 2);

  // ---- one-time pipeline ----
  cvt16_kernel<<<(Hd * Fd + 255) / 256, 256, 0, stream>>>(se_w1, w16_sew1, Hd * Fd);
  cvt16_kernel<<<(Hd * Hd + 255) / 256, 256, 0, stream>>>(se_w2, w16_sew2, Hd * Hd);
  cvt16_kernel<<<(H4d * H2d + 255) / 256, 256, 0, stream>>>(w_ih, w16_wih, H4d * H2d);
  cvt16_kernel<<<(Kd_ * Ad * Fd + 255) / 256, 256, 0, stream>>>(sub_w, w16_subw,
                                                                Kd_ * Ad * Fd);
  cvt16_kernel<<<(Hd * Hd + 255) / 256, 256, 0, stream>>>(hb1_w, w16_hb1, Hd * Hd);

  conv_kernel<<<(BTd * Fd) / 256, 256, 0, stream>>>(states, conv_w, conv_b, ws_conv);
  conv_kernel<<<(BTd * Fd) / 256, 256, 0, stream>>>(states, dconv_w, dconv_b, ws_dec);

  // hidden = relu(conv @ se_w1.T + se_b1)  : (BT x 128), K=576
  gemm_kernel<true, true><<<dim3(Hd / 64, BTd / 64), 128, 0, stream>>>(
      ws_conv, w16_sew1, se_b1, nullptr, ws_hidden, BTd, Hd, Fd, Hd);
  // s_emb = hidden @ se_w2.T + se_b2 -> emb[:, 0:128] (ldc = 256)
  gemm_kernel<false, true><<<dim3(Hd / 64, BTd / 64), 128, 0, stream>>>(
      ws_hidden, w16_sew2, se_b2, nullptr, ws_emb, BTd, Hd, Hd, H2d);
  // emb[:, 128:256] = act_emb[actions]
  act_gather_kernel<<<(BTd * Hd) / 256, 256, 0, stream>>>(actions, act_emb, ws_emb);
  // xproj = emb @ w_ih.T + b_ih + b_hh, stored fragment-major for the LSTM
  gemm_xproj_kernel<<<dim3(H4d / 64, (Td * 8) / 4), 128, 0, stream>>>(
      ws_emb, w16_wih, b_ih, b_hh, ws_xproj);
  // sub logits = dec @ sub_w(192x576).T + sub_b ; then softmax over A
  gemm_kernel<false, false><<<dim3((Kd_ * Ad) / 64, BTd / 64), 128, 0, stream>>>(
      ws_dec, w16_subw, sub_b, nullptr, ws_sublog, BTd, Kd_ * Ad, Fd, Kd_ * Ad);
  softmax12_kernel<<<(BTd * Kd_) / 256, 256, 0, stream>>>(ws_sublog);

  whh_tr_kernel<<<(Hd * H4d) / 256, 256, 0, stream>>>(w_hh, ws_whhT);
  init_mask_kernel<<<BTd / 256, 256, 0, stream>>>(ws_logcum, ws_mask);

  // ---- segment loop ----
  for (int seg = 0; seg < Md; ++seg) {
    float* enc_base = out + OUT_ENCS + (size_t)seg * BTd * Hd;
    lstm_kernel<<<Bd / 16, 256, 0, stream>>>(ws_xproj, ws_whhT, ws_mask, enc_base, ws_enc16);

    if (seg < Md - 1) {
      gemm_kernel<true, true><<<dim3(Hd / 64, BTd / 64), 128, 0, stream>>>(
          ws_enc16, w16_hb1, hb1_b, nullptr, ws_hid, BTd, Hd, Hd, Hd);
      float* lb = out + OUT_LB + (size_t)seg * BTd;
      logitsb_kernel<<<BTd / 256, 256, 0, stream>>>(ws_hid, hb2_w, hb2_b, lb);
      argmax_kernel<<<Bd, 256, 0, stream>>>(lb, ws_argmax);
    } else {
      lastidx_kernel<<<1, 128, 0, stream>>>(lengths, ws_argmax);
    }
    sb_kernel<<<BTd / 256, 256, 0, stream>>>(ws_argmax, out + OUT_SB + (size_t)seg * BTd);
    readout_kernel<<<Bd, Hd, 0, stream>>>(ws_argmax, enc_base, ws_ro);
    logitsz_kernel<<<Bd, Hd, 0, stream>>>(ws_ro, hz1_w, hz1_b, hz2_w, hz2_b,
                                          out + OUT_LZ + (size_t)seg * Bd * 2 * Kd_,
                                          out + OUT_SZ + (size_t)seg * Bd * Kd_);
    recs_kernel<<<(BTd * Ad) / 256, 256, 0, stream>>>(
        ws_sublog, out + OUT_SZ + (size_t)seg * Bd * Kd_,
        out + OUT_RECS + (size_t)seg * BTd * Ad);
    if (seg < Md - 1) {
      maskupd_kernel<<<BTd / 256, 256, 0, stream>>>(ws_argmax, ws_logcum, ws_mask,
                                                    out + OUT_MASKS + (size_t)seg * BTd);
    }
  }
}
